// NT_Xent_42769284334282
// MI455X (gfx1250) — matrix operand order repaired and checked by reading, not compiled, not run on hardware
//
#include <hip/hip_runtime.h>
#include <hip/hip_bf16.h>

// ---------------- types ----------------
typedef __bf16 bf16_t;
typedef bf16_t v16bf __attribute__((ext_vector_type(16)));
typedef float  v8f   __attribute__((ext_vector_type(8)));
typedef unsigned int uint4v __attribute__((ext_vector_type(4)));
typedef unsigned int uint8v __attribute__((ext_vector_type(8)));

#define B_ROWS 4096
#define K_DIM  1024
#define TAU_INV 2.0f      // 1 / 0.5

#define BLK       128     // block tile: 128 rows x 128 cols
#define KSTEP     64      // K per LDS stage
#define LDS_STRIDE 144    // 64 bf16 = 128B data + 16B pad (bank-conflict-free halves)
#define TILE_BYTES (BLK * LDS_STRIDE)   // 18 KB per tile

// ---------------- fp32 -> bf16 (round to nearest even) ----------------
__device__ __forceinline__ unsigned short f2bf(float f) {
    unsigned int u = __float_as_uint(f);
    u += 0x7FFFu + ((u >> 16) & 1u);
    return (unsigned short)(u >> 16);
}

// Kernel 1: convert Z to bf16 workspace, zero the neg accumulator.
__global__ __launch_bounds__(256) void ntxent_convert_zero(const float* __restrict__ Z,
                                                           unsigned short* __restrict__ Zb,
                                                           float* __restrict__ neg) {
    unsigned int tid    = blockIdx.x * blockDim.x + threadIdx.x;
    unsigned int stride = gridDim.x * blockDim.x;
    for (unsigned int i = tid; i < (unsigned int)B_ROWS * K_DIM; i += stride)
        Zb[i] = f2bf(Z[i]);
    if (tid < B_ROWS) neg[tid] = 0.0f;
}

// ---------------- CDNA5 async global->LDS copy (ASYNCcnt path) ----------------
// GV mode: vdst = LDS byte address, v[a:a+1] = 64-bit global address.
__device__ __forceinline__ void async_copy16(unsigned lds_byte, const char* gaddr) {
    asm volatile("global_load_async_to_lds_b128 %0, %1, off"
                 :: "v"(lds_byte), "v"(gaddr)
                 : "memory");
}
__device__ __forceinline__ void wait_async0() {
    asm volatile("s_wait_asynccnt 0x0" ::: "memory");
}

// Stage one 128x64 bf16 tile (global, row-major, row stride K_DIM) into padded LDS.
// 256 threads x 4 iterations x 16B = 16 KB of payload.
__device__ __forceinline__ void stage_tile(unsigned ldsBase, const char* gBase, int tid) {
#pragma unroll
    for (int i = 0; i < 4; ++i) {
        const int c   = tid + 256 * i;     // chunk id 0..1023
        const int r   = c >> 3;            // 8 x 16B chunks per row
        const int c16 = c & 7;
        async_copy16(ldsBase + (unsigned)(r * LDS_STRIDE + c16 * 16),
                     gBase + (size_t)r * (K_DIM * 2) + c16 * 16);
    }
}

// ---------------- LDS fragment loads ----------------
// A-matrix 16x32 bf16 (ISA 7.12.2): lanes 0-15: K 0..7 & 16..23; lanes 16-31: K 8..15 & 24..31.
__device__ __forceinline__ v16bf lds_frag_a(const char* base, int row0, int kk, int lane) {
    const int r  = row0 + (lane & 15);
    const int e0 = kk + ((lane >> 4) << 3);          // element offset: +0 or +8
    const char* p = base + r * LDS_STRIDE + e0 * 2;
    uint4v lo = *(const uint4v*)(p);                  // K = e0 .. e0+7
    uint4v hi = *(const uint4v*)(p + 32);             // K = e0+16 .. e0+23
    uint8v v;
    v[0] = lo[0]; v[1] = lo[1]; v[2] = lo[2]; v[3] = lo[3];
    v[4] = hi[0]; v[5] = hi[1]; v[6] = hi[2]; v[7] = hi[3];
    return __builtin_bit_cast(v16bf, v);
}
// B-matrix 32x16 bf16: lane n%16 = column n; lanes 0-15: K 0..7 & 16..23; lanes 16-31: K 8..15 & 24..31.
__device__ __forceinline__ v16bf lds_frag_b(const char* base, int row0, int kk, int lane) {
    const int r  = row0 + (lane & 15);
    const int e0 = kk + ((lane >> 4) << 3);          // element offset: +0 or +8
    const char* p = base + r * LDS_STRIDE + e0 * 2;
    uint4v lo = *(const uint4v*)(p);
    uint4v hi = *(const uint4v*)(p + 32);
    uint8v v;
    v[0] = lo[0]; v[1] = lo[1]; v[2] = lo[2]; v[3] = lo[3];
    v[4] = hi[0]; v[5] = hi[1]; v[6] = hi[2]; v[7] = hi[3];
    return __builtin_bit_cast(v16bf, v);
}

// Kernel 2: symmetric (upper-triangular) LDS double-buffered Z.Z^T with bf16 WMMA.
// sim is symmetric: each computed off-diagonal exp contributes to neg[row] AND neg[col],
// halving the GEMM work. 256 threads = 8 waves; block tile 128x128; wave tile 32x64.
__global__ __launch_bounds__(256) void ntxent_simgemm(const unsigned short* __restrict__ Zb,
                                                      float* __restrict__ neg) {
    if (blockIdx.y < blockIdx.x) return;             // upper-triangular tiles only

    __shared__ char smem[2][2][TILE_BYTES];          // [buffer][A/B][tile]: 72 KB

    const int tid  = threadIdx.x;
    const int lane = tid & 31;
    const int wave = tid >> 5;
    const int rowTile = blockIdx.x * BLK;
    const int colTile = blockIdx.y * BLK;
    const bool offDiag = (blockIdx.y != blockIdx.x);
    const int wRow = (wave & 3) * 32;                // wave's rows within tile
    const int wCol = (wave >> 2) * 64;               // wave's cols within tile

    const char* gA = (const char*)(Zb + (size_t)rowTile * K_DIM);
    const char* gB = (const char*)(Zb + (size_t)colTile * K_DIM);

    const v8f vzero = {0.f, 0.f, 0.f, 0.f, 0.f, 0.f, 0.f, 0.f};
    v8f acc[2][4];
#pragma unroll
    for (int m = 0; m < 2; ++m)
#pragma unroll
        for (int n = 0; n < 4; ++n)
            acc[m][n] = vzero;

    // Prologue: stage K-window 0 into buffer 0.
    stage_tile((unsigned)(uintptr_t)&smem[0][0][0], gA, tid);
    stage_tile((unsigned)(uintptr_t)&smem[0][1][0], gB, tid);
    wait_async0();
    __syncthreads();

    for (int kb = 0; kb < K_DIM; kb += KSTEP) {
        const int cur = (kb >> 6) & 1;
        if (kb + KSTEP < K_DIM) {                    // prefetch next window
            stage_tile((unsigned)(uintptr_t)&smem[cur ^ 1][0][0], gA + (size_t)(kb + KSTEP) * 2, tid);
            stage_tile((unsigned)(uintptr_t)&smem[cur ^ 1][1][0], gB + (size_t)(kb + KSTEP) * 2, tid);
        }

        const char* As = &smem[cur][0][0];
        const char* Bs = &smem[cur][1][0];
#pragma unroll
        for (int kk = 0; kk < KSTEP; kk += 32) {
            v16bf a0 = lds_frag_a(As, wRow,      kk, lane);
            v16bf a1 = lds_frag_a(As, wRow + 16, kk, lane);
            v16bf b[4];
#pragma unroll
            for (int n = 0; n < 4; ++n)
                b[n] = lds_frag_b(Bs, wCol + n * 16, kk, lane);
#pragma unroll
            for (int n = 0; n < 4; ++n) {
                acc[0][n] = __builtin_amdgcn_wmma_f32_16x16x32_bf16(
                    false, a0, false, b[n], (short)0, acc[0][n], false, false);
                acc[1][n] = __builtin_amdgcn_wmma_f32_16x16x32_bf16(
                    false, a1, false, b[n], (short)0, acc[1][n], false, false);
            }
        }

        wait_async0();       // our prefetch done
        __syncthreads();     // everyone done reading cur + all prefetch stores visible
    }

    // Epilogue: e = exp(sim/tau); diagonal masked (only diagonal tiles can hit it).
    // Row sums -> neg[row] always; column sums -> neg[col] for off-diagonal tiles
    // (symmetry: sim[r,c] == sim[c,r]).
    const int halfSel   = lane >> 4;
    const int colInTile = lane & 15;
    float colPartial[4] = {0.f, 0.f, 0.f, 0.f};
#pragma unroll
    for (int m = 0; m < 2; ++m) {
#pragma unroll
        for (int i = 0; i < 8; ++i) {
            const int row = rowTile + wRow + m * 16 + halfSel * 8 + i;
            float rowPartial = 0.0f;
#pragma unroll
            for (int n = 0; n < 4; ++n) {
                const int col = colTile + wCol + n * 16 + colInTile;
                float e = __expf(acc[m][n][i] * TAU_INV);
                e = (col == row) ? 0.0f : e;
                rowPartial += e;
                colPartial[n] += e;
            }
            rowPartial += __shfl_xor(rowPartial, 8);
            rowPartial += __shfl_xor(rowPartial, 4);
            rowPartial += __shfl_xor(rowPartial, 2);
            rowPartial += __shfl_xor(rowPartial, 1);
            if (colInTile == 0)
                atomicAdd(&neg[row], rowPartial);
        }
    }
    if (offDiag) {
#pragma unroll
        for (int n = 0; n < 4; ++n) {
            float cp = colPartial[n] + __shfl_xor(colPartial[n], 16);  // merge both 8-row halves
            if (lane < 16)
                atomicAdd(&neg[colTile + wCol + n * 16 + lane], cp);
        }
    }
}

// Kernel 3: positive-pair dots in fp32 (one thread per pair).
__global__ __launch_bounds__(256) void ntxent_pair(const float* __restrict__ Z,
                                                   float* __restrict__ pairv) {
    const int p = blockIdx.x * blockDim.x + threadIdx.x;
    if (p >= B_ROWS / 2) return;
    const float4* zi = (const float4*)(Z + (size_t)(2 * p) * K_DIM);
    const float4* zj = (const float4*)(Z + (size_t)(2 * p + 1) * K_DIM);
    float s = 0.f;
    for (int q = 0; q < K_DIM / 4; ++q) {
        float4 a = zi[q];
        float4 b = zj[q];
        s += a.x * b.x + a.y * b.y + a.z * b.z + a.w * b.w;
    }
    pairv[p] = s;
}

// Kernel 4: loss = (sum_b log(neg_b) - 2 * sum_k sim_k / tau) / B
__global__ __launch_bounds__(1024) void ntxent_finalize(const float* __restrict__ neg,
                                                        const float* __restrict__ pairv,
                                                        float* __restrict__ out) {
    __shared__ float red[1024];
    const int tid = threadIdx.x;
    float s = 0.f;
    for (int b = tid; b < B_ROWS; b += 1024) s += __logf(neg[b]);
    float sp = 0.f;
    for (int p = tid; p < B_ROWS / 2; p += 1024) sp += pairv[p];
    red[tid] = s - 2.0f * sp * TAU_INV;
    __syncthreads();
    for (int off = 512; off > 0; off >>= 1) {
        if (tid < off) red[tid] += red[tid + off];
        __syncthreads();
    }
    if (tid == 0) out[0] = red[0] / (float)B_ROWS;
}

extern "C" void kernel_launch(void* const* d_in, const int* in_sizes, int n_in,
                              void* d_out, int out_size, void* d_ws, size_t ws_size,
                              hipStream_t stream) {
    (void)in_sizes; (void)n_in; (void)out_size; (void)ws_size;
    const float* Z = (const float*)d_in[0];
    float* out = (float*)d_out;

    char* ws = (char*)d_ws;
    unsigned short* Zb    = (unsigned short*)ws;                          // 8 MB bf16
    float*          neg   = (float*)(ws + (size_t)B_ROWS * K_DIM * 2);    // 16 KB
    float*          pairv = neg + B_ROWS;                                 // 8 KB

    ntxent_convert_zero<<<4096, 256, 0, stream>>>(Z, Zb, neg);

    dim3 grid(B_ROWS / BLK, B_ROWS / BLK, 1);    // 32 x 32, upper triangle active
    ntxent_simgemm<<<grid, 256, 0, stream>>>(Zb, neg);

    ntxent_pair<<<(B_ROWS / 2) / 256, 256, 0, stream>>>(Z, pairv);

    ntxent_finalize<<<1, 1024, 0, stream>>>(neg, pairv, out);
}
